// Neurcomp_30863634989122
// MI455X (gfx1250) — matrix operand order repaired and checked by reading, not compiled, not run on hardware
//
#include <hip/hip_runtime.h>

typedef __attribute__((ext_vector_type(16))) _Float16 v16h;
typedef __attribute__((ext_vector_type(8)))  _Float16 v8h;
typedef __attribute__((ext_vector_type(8)))  float    v8f;
typedef __attribute__((ext_vector_type(4)))  int      v4i;
typedef __attribute__((ext_vector_type(4)))  unsigned int v4u;
typedef __attribute__((ext_vector_type(8)))  int      v8i;
typedef __attribute__((address_space(3)))    _Float16 lds_f16;

#define OMEGA   30.0f
#define HID     128
#define N_RES   7
#define ROWS    128      // points per block (4 waves x 32 rows)
#define THREADS 128
#define FRAGSZ  (32 * 16)          // halves per B fragment (one (ct,kc) tile)
#define CTSTRIDE (4 * FRAGSZ)      // halves between consecutive ct fragments

// ---------------------------------------------------------------------------
// Prep: convert res_w1/res_w2 (f32, row-major [o][h]) into f16 B-fragments.
// Layout: [m=layer*2+j][ct 0..7][kc 0..3][lane 0..31][t 0..15]
//   B[k][n] = W[n][k];  n = ct*16 + (lane&15);  k = kc*32 + (lane>=16?8:0)+t+(t>=8?8:0)
// Both wgt1 (0.5 for layers > 0, first matmul) and OMEGA are folded in, so
// the epilogue computes sin(acc + OMEGA*bias) with no extra multiply.
// ---------------------------------------------------------------------------
__global__ void neurcomp_prep(const float* __restrict__ rw1,
                              const float* __restrict__ rw2,
                              _Float16* __restrict__ wswz) {
    int idx = blockIdx.x * blockDim.x + threadIdx.x;
    const int total = N_RES * 2 * HID * HID;
    if (idx >= total) return;
    int t     = idx & 15;
    int lane  = (idx >> 4) & 31;
    int kc    = (idx >> 9) & 3;
    int ct    = (idx >> 11) & 7;
    int m     = idx >> 14;          // matmul index 0..13
    int layer = m >> 1;
    int j     = m & 1;
    int n = ct * 16 + (lane & 15);
    int k = kc * 32 + ((lane >> 4) ? 8 : 0) + t + ((t >= 8) ? 8 : 0);
    const float* W = (j == 0 ? rw1 : rw2) + layer * HID * HID;
    float scale = OMEGA * ((j == 0 && layer > 0) ? 0.5f : 1.0f);
    wswz[idx] = (_Float16)(scale * W[n * HID + k]);
}

// ---------------------------------------------------------------------------
// TDM: DMA one 32 KB weight matrix (4096 x 8B) from global into LDS.
// Tracked by TENSORcnt -> weight staging never touches DScnt/LOADcnt, so the
// WMMA loop's s_wait_dscnt only gates on its own A/B fragment loads.
// ---------------------------------------------------------------------------
__device__ __forceinline__ void tdm_load_weights(const _Float16* gsrc,
                                                 _Float16* ldst) {
    unsigned lds_off = (unsigned)(unsigned long long)(lds_f16*)ldst;
    unsigned long long ga = (unsigned long long)gsrc;
    v4u g0;
    g0[0] = 1u;                                   // count=1 (valid), user mode
    g0[1] = lds_off;                              // lds_addr (bytes)
    g0[2] = (unsigned)(ga & 0xFFFFFFFFu);         // global_addr[31:0]
    g0[3] = (unsigned)((ga >> 32) & 0x1FFFFFFu)   // global_addr[56:32]
          | (2u << 30);                           // type = 2 ("image")
    v8i g1;
    g1[0] = (int)(3u << 16);      // workgroup_mask=0, data_size=3 (8 bytes)
    g1[1] = (int)(4096u << 16);   // tensor_dim0[15:0] at bits 63:48
    g1[2] = (int)(1u << 16);      // tensor_dim0 hi=0 | tensor_dim1 lo16 = 1
    g1[3] = (int)(4096u << 16);   // tensor_dim1 hi=0 | tile_dim0 = 4096
    g1[4] = 1;                    // tile_dim1 = 1, tile_dim2 = 0
    g1[5] = 4096;                 // tensor_dim0_stride low 32
    g1[6] = 0;                    // stride0 hi | tensor_dim1_stride lo
    g1[7] = 0;                    // tensor_dim1_stride hi
    v4i g2 = {0, 0, 0, 0};        // <=2D tensor: groups 2/3 unused
    v4i g3 = {0, 0, 0, 0};
    v8i g4 = {0, 0, 0, 0, 0, 0, 0, 0};   // extra group (6-arg builtin form)
    __builtin_amdgcn_tensor_load_to_lds(g0, g1, g2, g3, g4, 0);
}

// ---------------------------------------------------------------------------
// Main fused SIREN forward. One block = 128 points, 4 waves x 32-row tiles.
// Each wave owns TWO 16-row M tiles so every B fragment feeds 2 WMMAs.
// Weight double-buffer in LDS is filled by the Tensor Data Mover, overlapped
// with the WMMA stream of the current matmul.
// ---------------------------------------------------------------------------
__global__ __launch_bounds__(THREADS) void neurcomp_siren(
    const float* __restrict__ x,
    const float* __restrict__ w0,
    const float* __restrict__ b0,
    const float* __restrict__ res_b1,
    const float* __restrict__ res_b2,
    const float* __restrict__ wf,
    const float* __restrict__ bf,
    const _Float16* __restrict__ wswz,
    float* __restrict__ out,
    int npts)
{
    __shared__ __attribute__((aligned(16))) _Float16 hA[ROWS * HID];    // 32 KB residual stream
    __shared__ __attribute__((aligned(16))) _Float16 sS[ROWS * HID];    // 32 KB s1 buffer
    __shared__ __attribute__((aligned(16))) _Float16 wB[2][HID * HID];  // 64 KB weight double-buffer
    __shared__ __attribute__((aligned(16))) float    xs[ROWS * 3];

    const int tid     = threadIdx.x;
    const int lane    = tid & 31;
    const int wave    = tid >> 5;
    const int rowbase = blockIdx.x * ROWS;

    // Kick off TDM for matmul 0's weights immediately (wave 0 issues once).
    if (wave == 0)
        tdm_load_weights(wswz, wB[0]);

    // Stage input coords
    for (int i = tid; i < ROWS * 3; i += THREADS) {
        int gr = rowbase + i / 3;
        xs[i] = (gr < npts) ? x[(size_t)rowbase * 3 + i] : 0.f;
    }
    __syncthreads();

    // ---- First sine layer: h = sin(omega * (x @ w0^T + b0)) ----
    for (int i = tid; i < ROWS * HID; i += THREADS) {
        int r = i >> 7, o = i & (HID - 1);
        float acc = b0[o]
                  + xs[r * 3 + 0] * w0[o * 3 + 0]
                  + xs[r * 3 + 1] * w0[o * 3 + 1]
                  + xs[r * 3 + 2] * w0[o * 3 + 2];
        hA[i] = (_Float16)__sinf(OMEGA * acc);
    }
    if (wave == 0)
        __builtin_amdgcn_s_wait_tensorcnt((short)0);
    __syncthreads();

    // A-fragment addressing (16-bit 16x32 A layout):
    //   lane<16 : row = lane,    K = [0..7],[16..23] of the chunk
    //   lane>=16: row = lane-16, K = [8..15],[24..31]
    const int arow0 = wave * 32 + (lane & 15);       // M tile 0
    const int arow1 = arow0 + 16;                    // M tile 1
    const int ahalf = (lane >> 4) * 8;

    #pragma unroll 1
    for (int m = 0; m < 2 * N_RES; ++m) {
        const int layer = m >> 1;
        const int j     = m & 1;
        const _Float16* actIn = (j == 0) ? hA : sS;
        const _Float16* wbuf  = wB[m & 1];

        // TDM-prefetch next weight matrix into the other buffer (overlaps WMMA)
        if (wave == 0 && m + 1 < 2 * N_RES)
            tdm_load_weights(wswz + (size_t)(m + 1) * HID * HID, wB[(m + 1) & 1]);

        v8f acc0[8], acc1[8];
        #pragma unroll
        for (int ct = 0; ct < 8; ++ct) { acc0[ct] = (v8f)0.f; acc1[ct] = (v8f)0.f; }

        #pragma unroll
        for (int kc = 0; kc < 4; ++kc) {
            const int koff = kc * 32 + ahalf;
            v8h a0lo = *(const v8h*)(actIn + arow0 * HID + koff);
            v8h a0hi = *(const v8h*)(actIn + arow0 * HID + koff + 16);
            v8h a1lo = *(const v8h*)(actIn + arow1 * HID + koff);
            v8h a1hi = *(const v8h*)(actIn + arow1 * HID + koff + 16);
            v16h a0, a1;
            #pragma unroll
            for (int t = 0; t < 8; ++t) {
                a0[t] = a0lo[t]; a0[t + 8] = a0hi[t];
                a1[t] = a1lo[t]; a1[t + 8] = a1hi[t];
            }

            // B double-buffer: loads for ct+1 are issued before the WMMAs of ct
            const _Float16* bbase = wbuf + kc * FRAGSZ + lane * 16;
            v8h blo = *(const v8h*)(bbase);
            v8h bhi = *(const v8h*)(bbase + 8);
            v8h bloN = blo, bhiN = bhi;
            #pragma unroll
            for (int ct = 0; ct < 8; ++ct) {
                if (ct < 7) {
                    const _Float16* bp = bbase + (ct + 1) * CTSTRIDE;
                    bloN = *(const v8h*)(bp);
                    bhiN = *(const v8h*)(bp + 8);
                }
                v16h b;
                #pragma unroll
                for (int t = 0; t < 8; ++t) { b[t] = blo[t]; b[t + 8] = bhi[t]; }
                acc0[ct] = __builtin_amdgcn_wmma_f32_16x16x32_f16(
                    false, a0, false, b, (short)0, acc0[ct], false, false);
                acc1[ct] = __builtin_amdgcn_wmma_f32_16x16x32_f16(
                    false, a1, false, b, (short)0, acc1[ct], false, false);
                blo = bloN; bhi = bhiN;
            }
        }

        // Epilogue on f32 accumulators. D layout: VGPR v, lane ->
        //   row = v + (lane<16 ? 0 : 8), col = ct*16 + (lane&15)
        // OMEGA already folded into weights; bias scaled by OMEGA here.
        const float* bias = ((j == 0) ? res_b1 : res_b2) + layer * HID;
        const int rbase0 = wave * 32 + ((lane >> 4) ? 8 : 0);
        if (j == 0) {
            #pragma unroll
            for (int ct = 0; ct < 8; ++ct) {
                const int col = ct * 16 + (lane & 15);
                const float bv = OMEGA * bias[col];
                #pragma unroll
                for (int v = 0; v < 8; ++v) {
                    sS[(rbase0 + v) * HID + col] =
                        (_Float16)__sinf(acc0[ct][v] + bv);
                    sS[(rbase0 + 16 + v) * HID + col] =
                        (_Float16)__sinf(acc1[ct][v] + bv);
                }
            }
        } else {
            const float wgt2 = (layer == N_RES - 1) ? 0.5f : 1.0f;
            #pragma unroll
            for (int ct = 0; ct < 8; ++ct) {
                const int col = ct * 16 + (lane & 15);
                const float bv = OMEGA * bias[col];
                #pragma unroll
                for (int v = 0; v < 8; ++v) {
                    int r0 = (rbase0 + v) * HID + col;
                    int r1 = (rbase0 + 16 + v) * HID + col;
                    float s20 = __sinf(acc0[ct][v] + bv);
                    float s21 = __sinf(acc1[ct][v] + bv);
                    hA[r0] = (_Float16)(wgt2 * ((float)hA[r0] + s20));
                    hA[r1] = (_Float16)(wgt2 * ((float)hA[r1] + s21));
                }
            }
        }
        // Wave 0 drains its TDM transfer; barrier makes it visible to all.
        if (wave == 0)
            __builtin_amdgcn_s_wait_tensorcnt((short)0);
        __syncthreads();
    }

    // ---- Final linear: out = h @ wf^T + bf (D_OUT = 1) ----
    {
        int gr = rowbase + tid;
        if (gr < npts) {
            float sum = bf[0];
            #pragma unroll 8
            for (int k = 0; k < HID; ++k)
                sum += (float)hA[tid * HID + k] * wf[k];
            out[gr] = sum;
        }
    }
}

extern "C" void kernel_launch(void* const* d_in, const int* in_sizes, int n_in,
                              void* d_out, int out_size, void* d_ws, size_t ws_size,
                              hipStream_t stream) {
    const float* x   = (const float*)d_in[0];
    const float* w0  = (const float*)d_in[1];
    const float* b0  = (const float*)d_in[2];
    const float* rw1 = (const float*)d_in[3];
    const float* rb1 = (const float*)d_in[4];
    const float* rw2 = (const float*)d_in[5];
    const float* rb2 = (const float*)d_in[6];
    const float* wf  = (const float*)d_in[7];
    const float* bf  = (const float*)d_in[8];
    float* out = (float*)d_out;
    _Float16* wswz = (_Float16*)d_ws;   // 14 * 16384 halves = 448 KB swizzled f16 weights

    int npts = in_sizes[0] / 3;

    const int prep_total = N_RES * 2 * HID * HID;
    neurcomp_prep<<<(prep_total + 255) / 256, 256, 0, stream>>>(rw1, rw2, wswz);

    int nblocks = (npts + ROWS - 1) / ROWS;
    neurcomp_siren<<<nblocks, THREADS, 0, stream>>>(x, w0, b0, rb1, rb2, wf, bf,
                                                    wswz, out, npts);
}
